// DeltaModel_3204045603476
// MI455X (gfx1250) — compile-verified, matrix operand order checked by reading, not run on hardware
//
#include <hip/hip_runtime.h>
#include <hip/hip_bf16.h>

// Problem constants (match reference)
#define B_  256
#define L_  4096
#define H_  64
#define V_  64
#define LN_EPS 1e-5f

typedef __attribute__((ext_vector_type(16))) _Float16 v16h;
typedef __attribute__((ext_vector_type(8)))  float    v8f;
typedef __attribute__((ext_vector_type(2)))  float    v2f;

// ---------------------------------------------------------------------------
// WMMA fragment helpers, f16 16x16x32 (ISA 7.12.2):
//  A (16x32, MxK): lane -> row r = lane&15, kb = 8*(lane>>4);
//                  a[j]   = A[r][kk+kb+j],  a[j+8] = A[r][kk+16+kb+j]
//  B (32x16, KxN): symmetric with N in place of M (col n = lane&15)
//  C/D (16x16 f32): c[i] = C[i + 8*(lane>>4)][lane&15]
// f32 16x16x4:
//  A (16x4): lane -> row r = lane&15, kp = 2*(lane>>4); a = A[r][kk+kp .. +1]
//  B (4x16): mirror with col n = lane&15
// ---------------------------------------------------------------------------
__device__ __forceinline__ v16h frag_a_f32row(const float* __restrict__ row,
                                              int kk, int kb) {
  v16h a;
#pragma unroll
  for (int j = 0; j < 8; ++j) {
    a[j]     = (_Float16)row[kk + kb + j];
    a[j + 8] = (_Float16)row[kk + 16 + kb + j];
  }
  return a;
}

__device__ __forceinline__ v16h frag_b_f32(const float* __restrict__ W, int ld,
                                           int kk, int kb, int ncol) {
  v16h b;
#pragma unroll
  for (int j = 0; j < 8; ++j) {
    b[j]     = (_Float16)W[(kk + kb + j) * ld + ncol];
    b[j + 8] = (_Float16)W[(kk + 16 + kb + j) * ld + ncol];
  }
  return b;
}

// ---------------------------------------------------------------------------
// Kernel A: h = LayerNorm(embed[seq] + FFN(embed[seq]))  -> hout (f32, [B*L,64])
// One wave per block; v_wmma_f32_16x16x32_f16 for both GEMMs.
// ---------------------------------------------------------------------------
__global__ __launch_bounds__(32) void ffn_ln_kernel(
    const int*   __restrict__ seq,
    const float* __restrict__ embed,   // [V,64]
    const float* __restrict__ w1,      // [64,128]
    const float* __restrict__ b1,      // [128]
    const float* __restrict__ w2,      // [128,64]
    const float* __restrict__ b2,      // [64]
    const float* __restrict__ lnw,     // [64]
    const float* __restrict__ lnb,     // [64]
    float*       __restrict__ hout)    // [B*L,64]
{
  __shared__ _Float16 h1s[16 * 128];
  __shared__ float    xs[16 * 64];

  const int lane = threadIdx.x;
  const int r    = lane & 15;
  const int hi   = lane >> 4;
  const int kb   = 8 * hi;

  v16h w1f[8][2];
#pragma unroll
  for (int n = 0; n < 8; ++n)
#pragma unroll
    for (int k = 0; k < 2; ++k)
      w1f[n][k] = frag_b_f32(w1, 128, 32 * k, kb, 16 * n + r);

  v16h w2f[4][4];
#pragma unroll
  for (int n = 0; n < 4; ++n)
#pragma unroll
    for (int k = 0; k < 4; ++k)
      w2f[n][k] = frag_b_f32(w2, 64, 32 * k, kb, 16 * n + r);

  const int ntiles = (B_ * L_) / 16;
  for (int tile = blockIdx.x; tile < ntiles; tile += gridDim.x) {
    const int tok0 = tile * 16;

    const int    id_r = seq[tok0 + r];
    const float* erow = embed + id_r * H_;
    const v16h a0 = frag_a_f32row(erow, 0,  kb);
    const v16h a1 = frag_a_f32row(erow, 32, kb);

    // layer 1 + ReLU -> LDS (f16)
#pragma unroll
    for (int n = 0; n < 8; ++n) {
      v8f acc = {};
      acc = __builtin_amdgcn_wmma_f32_16x16x32_f16(false, a0, false, w1f[n][0],
                                                   (short)0, acc, false, false);
      acc = __builtin_amdgcn_wmma_f32_16x16x32_f16(false, a1, false, w1f[n][1],
                                                   (short)0, acc, false, false);
      const int   nc   = 16 * n + r;
      const float bias = b1[nc];
#pragma unroll
      for (int i = 0; i < 8; ++i) {
        float v = acc[i] + bias;
        v = v > 0.f ? v : 0.f;
        h1s[(i + 8 * hi) * 128 + nc] = (_Float16)v;
      }
    }
    __syncthreads();

    // layer 2 + bias + residual -> xs
    v16h a2[4];
#pragma unroll
    for (int k = 0; k < 4; ++k) {
      v16h a;
#pragma unroll
      for (int j = 0; j < 8; ++j) {
        a[j]     = h1s[r * 128 + 32 * k + kb + j];
        a[j + 8] = h1s[r * 128 + 32 * k + 16 + kb + j];
      }
      a2[k] = a;
    }
#pragma unroll
    for (int n = 0; n < 4; ++n) {
      v8f acc = {};
#pragma unroll
      for (int k = 0; k < 4; ++k)
        acc = __builtin_amdgcn_wmma_f32_16x16x32_f16(false, a2[k], false,
                                                     w2f[n][k], (short)0, acc,
                                                     false, false);
      const int   nc   = 16 * n + r;
      const float bias = b2[nc];
#pragma unroll
      for (int i = 0; i < 8; ++i) {
        const int m   = i + 8 * hi;
        const int idm = seq[tok0 + m];
        xs[m * 64 + nc] = acc[i] + bias + embed[idm * H_ + nc];
      }
    }
    __syncthreads();

    // LayerNorm per token (lanes 0..15)
    if (lane < 16) {
      const float* x = &xs[lane * 64];
      float mu = 0.f;
#pragma unroll
      for (int j = 0; j < 64; ++j) mu += x[j];
      mu *= (1.0f / 64.0f);
      float var = 0.f;
#pragma unroll
      for (int j = 0; j < 64; ++j) { const float d = x[j] - mu; var += d * d; }
      var *= (1.0f / 64.0f);
      const float rs = rsqrtf(var + LN_EPS);
      float* dst = hout + (size_t)(tok0 + lane) * 64;
#pragma unroll
      for (int j = 0; j < 64; ++j)
        dst[j] = (x[j] - mu) * rs * lnw[j] + lnb[j];
    }
    __syncthreads();
  }
}

// ---------------------------------------------------------------------------
// Async staging (GLOBAL_LOAD_ASYNC_TO_LDS_B128, ASYNCcnt)
// ---------------------------------------------------------------------------
__device__ __forceinline__ void async_copy_row_b128(const float* gsrc,
                                                    float* ldst) {
  const unsigned l = (unsigned)(uintptr_t)ldst;
#pragma unroll
  for (int j = 0; j < 16; ++j) {
    asm volatile("global_load_async_to_lds_b128 %0, %1, off"
                 :
                 : "v"(l + 16u * j), "v"(gsrc + 4 * j)
                 : "memory");
  }
}

__device__ __forceinline__ void wait_async0() {
  asm volatile("s_wait_asynccnt 0x0" ::: "memory");
}

// ---------------------------------------------------------------------------
// Kernel B: chunked delta-rule scan (chunk C=16) on f32 WMMA + readout.
// One block (2 waves) per batch. With beta_t = 1/(k_t.k_t+1e-6):
//   S  = K K^T                      (Gram, per-wave, 16 WMMA)
//   P  = K M0^T ; g_t = k_t - beta_t p_t   (per wave: its 2 n-tiles, 32 WMMA)
//   dv_t = g_t - beta_t Sum_{s<t} S[t,s] dv_s   (forward substitution)
//   M += DV^T K                     (per wave: its 2 m-tiles, 32 WMMA)
// M rows are wave-partitioned (wave w owns rows [32w,32w+32)): P reads, the
// update writes, and the readout touch only own rows, so each chunk needs a
// single barrier (publishing the async-staged next K chunk). Zero-padded tail
// rows contribute nothing (S row = 0, g = 0 => dv = 0).
// ---------------------------------------------------------------------------
__global__ __launch_bounds__(64) void scan_chunk_kernel(
    const float* __restrict__ h,    // [B,L,64]
    const float* __restrict__ rw,   // [64,64]
    const float* __restrict__ rb,   // [64]
    const float* __restrict__ ow,   // [64,64]
    const float* __restrict__ ob,   // [64]
    float*       __restrict__ out)  // [B,64]
{
  __shared__ float Mlds[64 * 64];      // state, row-major
  __shared__ float Klds[2][16 * 64];   // double-buffered K chunk
  __shared__ float Glds[16 * 64];      // g_t rows (column-partitioned by wave)
  __shared__ float DVlds[16 * 64];     // dv_t rows (column-partitioned by wave)
  __shared__ float Slds[2][16 * 16];   // per-wave Gram
  __shared__ float betal[2][16];       // per-wave 1/(k.k+eps)
  __shared__ float red[64];

  const int tid = threadIdx.x;
  const int w   = tid >> 5;            // wave id (0/1)
  const int wl  = tid & 31;            // lane in wave
  const int nn  = wl & 15;             // fragment row/col index
  const int hi  = wl >> 4;             // half-wave
  const int kp  = 2 * hi;              // f32-frag K-pair base
  const int b   = blockIdx.x;
  const float* hb = h + (size_t)b * (size_t)L_ * H_;

  // zero own M rows (wave-local)
  {
    float4* mrow = (float4*)&Mlds[tid * 64];
#pragma unroll
    for (int j = 0; j < 16; ++j) mrow[j] = make_float4(0.f, 0.f, 0.f, 0.f);
  }

  const int nsteps  = L_ - 1;              // 4095
  const int nchunks = (nsteps + 15) / 16;  // 256

  // Prologue: stage chunk 0.
  if (tid < 16) {
    if (tid < min(16, nsteps))
      async_copy_row_b128(hb + (size_t)tid * H_, &Klds[0][tid * 64]);
  }
  wait_async0();
  __syncthreads();

  for (int c = 0; c < nchunks; ++c) {
    const int cur = c & 1;
    const float* Kc = Klds[cur];

    // Stream next chunk (wave0 lanes); zero-pad missing tail rows.
    if (c + 1 < nchunks && tid < 16) {
      const int t1  = (c + 1) * 16;
      const int nt1 = min(16, nsteps - t1);
      if (tid < nt1) {
        async_copy_row_b128(hb + (size_t)(t1 + tid) * H_,
                            &Klds[cur ^ 1][tid * 64]);
      } else {
        float4* z = (float4*)&Klds[cur ^ 1][tid * 64];
#pragma unroll
        for (int j = 0; j < 16; ++j) z[j] = make_float4(0.f, 0.f, 0.f, 0.f);
      }
    }

    // Per-wave beta (same-wave LDS ordering => no barrier needed).
    if (wl < 16) {
      const float4* row = (const float4*)&Kc[wl * 64];
      float ss = 0.f;
#pragma unroll
      for (int j = 0; j < 16; ++j) {
        const float4 v = row[j];
        ss += v.x * v.x + v.y * v.y + v.z * v.z + v.w * v.w;
      }
      betal[w][wl] = 1.0f / (ss + 1e-6f);
    }

    // K fragments: serve as A (rows) AND Gram-B (K^T cols) since r==n==nn.
    v2f kf[16];
#pragma unroll
    for (int kk = 0; kk < 16; ++kk)
      kf[kk] = *(const v2f*)&Kc[nn * 64 + 4 * kk + kp];

    // ---- Gram: S = K K^T ----
    {
      v8f s = {};
#pragma unroll
      for (int kk = 0; kk < 16; ++kk)
        s = __builtin_amdgcn_wmma_f32_16x16x4_f32(false, kf[kk], false, kf[kk],
                                                  (short)0, s, false, false);
#pragma unroll
      for (int i = 0; i < 8; ++i) Slds[w][(i + 8 * hi) * 16 + nn] = s[i];
    }

    // ---- P = K M0^T on this wave's two n-tiles; G = K - beta*P ----
#pragma unroll
    for (int q2 = 0; q2 < 2; ++q2) {
      const int ni = 2 * w + q2;
      v8f p = {};
#pragma unroll
      for (int kk = 0; kk < 16; ++kk) {
        const v2f bm = *(const v2f*)&Mlds[(16 * ni + nn) * 64 + 4 * kk + kp];
        p = __builtin_amdgcn_wmma_f32_16x16x4_f32(false, kf[kk], false, bm,
                                                  (short)0, p, false, false);
      }
#pragma unroll
      for (int i = 0; i < 8; ++i) {
        const int t   = i + 8 * hi;
        const int col = 16 * ni + nn;
        Glds[t * 64 + col] = Kc[t * 64 + col] - betal[w][t] * p[i];
      }
    }

    // ---- Forward substitution (lane owns column c0 = 32w + wl) ----
    {
      const int c0 = 32 * w + wl;
      float dv[16], g[16];
#pragma unroll
      for (int t = 0; t < 16; ++t) g[t] = Glds[t * 64 + c0];
#pragma unroll
      for (int t = 0; t < 16; ++t) {
        float acc = 0.f;
#pragma unroll
        for (int s = 0; s < t; ++s)
          acc += Slds[w][t * 16 + s] * dv[s];
        dv[t] = g[t] - betal[w][t] * acc;
        DVlds[t * 64 + c0] = dv[t];
      }
    }

    // ---- Update: M += DV^T K on this wave's two m-tiles ----
#pragma unroll
    for (int m2 = 0; m2 < 2; ++m2) {
      const int mi = 2 * w + m2;
      v2f aD[4];
#pragma unroll
      for (int kk = 0; kk < 4; ++kk) {
        v2f a;
        a[0] = DVlds[(4 * kk + kp) * 64 + 16 * mi + nn];
        a[1] = DVlds[(4 * kk + kp + 1) * 64 + 16 * mi + nn];
        aD[kk] = a;
      }
#pragma unroll
      for (int ni = 0; ni < 4; ++ni) {
        v8f cf;
#pragma unroll
        for (int i = 0; i < 8; ++i)
          cf[i] = Mlds[(16 * mi + i + 8 * hi) * 64 + 16 * ni + nn];
#pragma unroll
        for (int kk = 0; kk < 4; ++kk) {
          v2f bv;
          bv[0] = Kc[(4 * kk + kp) * 64 + 16 * ni + nn];
          bv[1] = Kc[(4 * kk + kp + 1) * 64 + 16 * ni + nn];
          cf = __builtin_amdgcn_wmma_f32_16x16x4_f32(false, aD[kk], false, bv,
                                                     (short)0, cf, false, false);
        }
#pragma unroll
        for (int i = 0; i < 8; ++i)
          Mlds[(16 * mi + i + 8 * hi) * 64 + 16 * ni + nn] = cf[i];
      }
    }

    // Publish the freshly streamed K buffer; next chunk reads own M rows only.
    wait_async0();
    __syncthreads();
  }

  // ---- ctx = M.q (thread tid owns row tid; rows are own-wave data) ----
  const float* qptr = hb + (size_t)(L_ - 1) * H_;
  {
    const float4* mr = (const float4*)&Mlds[tid * 64];
    const float4* q4 = (const float4*)qptr;
    float c0 = 0.f, c1 = 0.f;
#pragma unroll
    for (int j = 0; j < 16; ++j) {
      const float4 m = mr[j], qq = q4[j];
      c0 += m.x * qq.x + m.y * qq.y;
      c1 += m.z * qq.z + m.w * qq.w;
    }
    red[tid] = c0 + c1;
  }
  __syncthreads();

  // r = ctx @ read_w + read_b
  float rv = rb[tid];
#pragma unroll 8
  for (int j = 0; j < 64; ++j) rv += red[j] * rw[j * 64 + tid];
  __syncthreads();
  red[tid] = rv;
  __syncthreads();

  // out = r @ out_w + out_b
  float o = ob[tid];
#pragma unroll 8
  for (int j = 0; j < 64; ++j) o += red[j] * ow[j * 64 + tid];
  out[b * 64 + tid] = o;
}

// ---------------------------------------------------------------------------
extern "C" void kernel_launch(void* const* d_in, const int* in_sizes, int n_in,
                              void* d_out, int out_size, void* d_ws, size_t ws_size,
                              hipStream_t stream) {
  const int*   seq   = (const int*)  d_in[0];
  const float* embed = (const float*)d_in[1];
  const float* ff_w1 = (const float*)d_in[2];
  const float* ff_b1 = (const float*)d_in[3];
  const float* ff_w2 = (const float*)d_in[4];
  const float* ff_b2 = (const float*)d_in[5];
  const float* ln_w  = (const float*)d_in[6];
  const float* ln_b  = (const float*)d_in[7];
  const float* rd_w  = (const float*)d_in[8];
  const float* rd_b  = (const float*)d_in[9];
  const float* out_w = (const float*)d_in[10];
  const float* out_b = (const float*)d_in[11];

  float* hbuf = (float*)d_ws;  // [B*L, 64] f32 = 256 MiB scratch

  ffn_ln_kernel<<<2048, 32, 0, stream>>>(seq, embed, ff_w1, ff_b1, ff_w2, ff_b2,
                                         ln_w, ln_b, hbuf);
  scan_chunk_kernel<<<B_, 64, 0, stream>>>(hbuf, rd_w, rd_b, out_w, out_b,
                                           (float*)d_out);
}
